// AFTFull_1451698946440
// MI455X (gfx1250) — matrix operand order repaired and checked by reading, not compiled, not run on hardware
//
#include <hip/hip_runtime.h>
#include <hip/hip_bf16.h>
#include <hip/hip_fp16.h>

typedef __attribute__((ext_vector_type(16))) _Float16 v16h;
typedef __attribute__((ext_vector_type(8)))  float    v8f;

#define BSZ 8
#define TSZ 4096
#define DSZ 512
#define HSZ 64
#define MSZ (BSZ * TSZ)   // 32768
#define APAD 40           // LDS row stride in halfs (32 data + 8 pad) -> 80B, keeps 16B alignment
#define ZLP 136           // epilogue transpose row stride (128 + 8)

// sched_group_barrier class masks
#define SG_MFMA   0x008
#define SG_VMREAD 0x020
#define SG_DSREAD 0x100
#define SG_DSWRITE 0x200
#define SGB(m, n) __builtin_amdgcn_sched_group_barrier((m), (n), 0)

union FragH { v16h h; unsigned u[8]; };

// A fragment: 16x32 f16, rows M=mBase..mBase+15 from row-major LDS tile (stride APAD).
// ISA layout: lanes 0-15 / 16-31 both hold M=lane%16; VGPR v holds K pair at
// K = (v/4)*16 + (lane/16)*8 + (v%4)*2.  (vectorizes to 2x ds_load_b128)
__device__ __forceinline__ v16h load_frag_a(const _Float16* lds, int mBase, int lane) {
  FragH f;
  const int g = (lane >> 4) & 1, mr = lane & 15;
  const _Float16* row = lds + (mBase + mr) * APAD;
#pragma unroll
  for (int v = 0; v < 8; ++v) {
    int k = ((v >> 2) << 4) + (g << 3) + ((v & 3) << 1);
    f.u[v] = *reinterpret_cast<const unsigned*>(row + k);
  }
  return f.h;
}

// B fragment: 32x16 f16 (KxN), from LDS stored [n][k] (stride APAD).
// B striping (matches A): VGPR v holds K pair at
// K = (v/4)*16 + (lane/16)*8 + (v%4)*2; N = lane%16.  (vectorizes to 2x ds_load_b128)
__device__ __forceinline__ v16h load_frag_b(const _Float16* lds, int nBase, int lane) {
  FragH f;
  const int g = (lane >> 4) & 1, nr = lane & 15;
  const _Float16* col = lds + (nBase + nr) * APAD + (g << 3);
#pragma unroll
  for (int v = 0; v < 8; ++v)
    f.u[v] = *reinterpret_cast<const unsigned*>(col + ((v >> 2) << 4) + ((v & 3) << 1));
  return f.h;
}

// ---------------- prep kernels (vectorized: float4 in, 4xf16 out) ----------------
union H4 { uint2 u; _Float16 h[4]; };

__global__ void k_cast_f32_f16(const float* __restrict__ src, _Float16* __restrict__ dst, int n4) {
  int i = blockIdx.x * blockDim.x + threadIdx.x;
  int stride = gridDim.x * blockDim.x;
  for (; i < n4; i += stride) {
    float4 s = reinterpret_cast<const float4*>(src)[i];
    H4 o;
    o.h[0] = (_Float16)s.x; o.h[1] = (_Float16)s.y; o.h[2] = (_Float16)s.z; o.h[3] = (_Float16)s.w;
    reinterpret_cast<uint2*>(dst)[i] = o.u;
  }
}

__global__ void k_exp_f32_f16(const float* __restrict__ src, _Float16* __restrict__ dst, int n4) {
  int i = blockIdx.x * blockDim.x + threadIdx.x;
  int stride = gridDim.x * blockDim.x;
  for (; i < n4; i += stride) {
    float4 s = reinterpret_cast<const float4*>(src)[i];
    H4 o;
    o.h[0] = (_Float16)expf(s.x); o.h[1] = (_Float16)expf(s.y);
    o.h[2] = (_Float16)expf(s.z); o.h[3] = (_Float16)expf(s.w);
    reinterpret_cast<uint2*>(dst)[i] = o.u;
  }
}

// ---------------- GEMM 1: QKV projection + fused epilogue ----------------
// C[32768 x 192] = xh[32768 x 512] @ wqkvT'[512 x 192] (wqkvT stored [192][512] row-major).
// Epilogue -> sigQ (f32, [B*T][64]), Zt (f16, [B*128 h'][T]): h'<64 holds eK*V, h'>=64 holds eK.
__global__ __launch_bounds__(256) void k_qkv_gemm(
    const _Float16* __restrict__ xh, const _Float16* __restrict__ wqkvT,
    const float* __restrict__ bq, const float* __restrict__ bk, const float* __restrict__ bv,
    float* __restrict__ sigQ, _Float16* __restrict__ Zt) {
  __shared__ _Float16 smem[2 * 128 * APAD + 2 * 192 * APAD];  // 25600 halfs; Zl (17408) overlays later
  _Float16* As = smem;
  _Float16* Bs = smem + 2 * 128 * APAD;
  const int tid = threadIdx.x, wave = tid >> 5, lane = tid & 31;
  const int M0 = blockIdx.x * 128;
  const int arow = tid >> 1, ap16 = (tid & 1) << 4;
  const int brow = (tid < 192) ? tid : 191;   // branchless: extra threads duplicate row 191
  v8f acc[12] = {};

  uint4 sa0, sa1, sb0, sb1, sb2, sb3;
  auto ldg = [&](int K0) {
    const uint4* pa = reinterpret_cast<const uint4*>(xh + (size_t)(M0 + arow) * DSZ + K0 + ap16);
    sa0 = pa[0]; sa1 = pa[1];
    const uint4* pb = reinterpret_cast<const uint4*>(wqkvT + (size_t)brow * DSZ + K0);
    sb0 = pb[0]; sb1 = pb[1]; sb2 = pb[2]; sb3 = pb[3];
  };
  auto sts = [&](int nb) {
    uint4* da = reinterpret_cast<uint4*>(&As[nb * 128 * APAD + arow * APAD + ap16]);
    da[0] = sa0; da[1] = sa1;
    uint4* db = reinterpret_cast<uint4*>(&Bs[nb * 192 * APAD + brow * APAD]);
    db[0] = sb0; db[1] = sb1; db[2] = sb2; db[3] = sb3;
  };

  ldg(0); sts(0);
  __syncthreads();
  int buf = 0;
  const int NIT = DSZ / 32;
  for (int it = 0; it < NIT; ++it) {
    const int itn = (it + 1 < NIT) ? (it + 1) : it;   // branchless double-buffer (last re-stages dead tile)
    ldg(itn * 32);
    const _Float16* Ab = &As[buf * 128 * APAD];
    const _Float16* Bb = &Bs[buf * 192 * APAD];
    v16h a = load_frag_a(Ab, wave * 16, lane);
#pragma unroll
    for (int grp = 0; grp < 3; ++grp) {
      v16h bf[4];
#pragma unroll
      for (int j = 0; j < 4; ++j) bf[j] = load_frag_b(Bb, (grp * 4 + j) * 16, lane);
#pragma unroll
      for (int j = 0; j < 4; ++j)
        acc[grp * 4 + j] = __builtin_amdgcn_wmma_f32_16x16x32_f16(false, a, false, bf[j], (short)0,
                                                                  acc[grp * 4 + j], false, false);
    }
    sts(buf ^ 1);
    // pipeline: global reads -> all frag reads -> wmma/ds-write overlap
    SGB(SG_VMREAD, 6);
    SGB(SG_DSREAD, 26);
    SGB(SG_MFMA, 6);
    SGB(SG_DSWRITE, 6);
    SGB(SG_MFMA, 6);
    __syncthreads();
    buf ^= 1;
  }

  // epilogue: compute sigQ / eK*V / eK; transpose Z through LDS into Zt[b*128+h'][T]
  const int g = lane >> 4, nr = lane & 15;
  const int bb = blockIdx.x >> 5;           // batch index (4096/128 = 32 blocks per batch)
  const int t0 = (blockIdx.x & 31) * 128;   // t-tile base within batch
  const int rowl_base = wave * 16 + g * 8;
  _Float16* Zl = smem;                      // reuse tile memory (all compute reads done)
#pragma unroll
  for (int nt = 0; nt < 4; ++nt) {
    const int h = nt * 16 + nr;
    const float bqv = bq[h], bkv = bk[h], bvv = bv[h];
#pragma unroll
    for (int r = 0; r < 8; ++r) {
      const int rowl = rowl_base + r;
      const int grow = M0 + rowl;
      float q  = acc[nt][r] + bqv;
      float kk = acc[nt + 4][r] + bkv;
      float vv = acc[nt + 8][r] + bvv;
      float sq = 1.0f / (1.0f + expf(-q));
      float ek = expf(kk);
      sigQ[(size_t)grow * HSZ + h] = sq;
      Zl[h * ZLP + rowl]        = (_Float16)(ek * vv);
      Zl[(64 + h) * ZLP + rowl] = (_Float16)ek;
    }
  }
  __syncthreads();
  {
    const int hp = tid >> 1, seg = (tid & 1) << 6;   // 128 h' rows x 2 segments of 64 t
    const size_t gbase = ((size_t)bb * 128 + hp) * TSZ + t0 + seg;
#pragma unroll
    for (int v = 0; v < 8; ++v) {
      uint4 val = *reinterpret_cast<const uint4*>(&Zl[hp * ZLP + seg + v * 8]);
      *reinterpret_cast<uint4*>(Zt + gbase + v * 8) = val;
    }
  }
}

// ---------------- GEMM 2: the AFT core, per batch ----------------
// [num|den][4096 x 128] = ew[4096 x 4096] @ Zt_b^T; epilogue: Yt = sigQ * num/den (f16)
__global__ __launch_bounds__(256) void k_aft_core(
    const _Float16* __restrict__ ewh, const _Float16* __restrict__ Zt,
    const float* __restrict__ sigQ, _Float16* __restrict__ Yth) {
  __shared__ _Float16 As[2 * 128 * APAD];
  __shared__ _Float16 Bs[2 * 128 * APAD];
  const int tid = threadIdx.x, wave = tid >> 5, lane = tid & 31;
  const int M0 = blockIdx.x * 128;
  const int b = blockIdx.y;
  const int arow = tid >> 1, ap16 = (tid & 1) << 4;
  const _Float16* zrow = Zt + ((size_t)b * 128 + arow) * TSZ + ap16;
  const _Float16* erow = ewh + (size_t)(M0 + arow) * TSZ + ap16;
  v8f acc[8] = {};

  uint4 sa0, sa1, sb0, sb1;
  auto ldg = [&](int K0) {
    const uint4* pa = reinterpret_cast<const uint4*>(erow + K0);
    sa0 = pa[0]; sa1 = pa[1];
    const uint4* pb = reinterpret_cast<const uint4*>(zrow + K0);
    sb0 = pb[0]; sb1 = pb[1];
  };
  auto sts = [&](int nb) {
    uint4* da = reinterpret_cast<uint4*>(&As[nb * 128 * APAD + arow * APAD + ap16]);
    da[0] = sa0; da[1] = sa1;
    uint4* db = reinterpret_cast<uint4*>(&Bs[nb * 128 * APAD + arow * APAD + ap16]);
    db[0] = sb0; db[1] = sb1;
  };

  ldg(0); sts(0);
  __syncthreads();
  int buf = 0;
  const int NIT = TSZ / 32;
  for (int it = 0; it < NIT; ++it) {
    const int itn = (it + 1 < NIT) ? (it + 1) : it;
    ldg(itn * 32);
    const _Float16* Ab = &As[buf * 128 * APAD];
    const _Float16* Bb = &Bs[buf * 128 * APAD];
    v16h a = load_frag_a(Ab, wave * 16, lane);
#pragma unroll
    for (int grp = 0; grp < 2; ++grp) {
      v16h bf[4];
#pragma unroll
      for (int j = 0; j < 4; ++j) bf[j] = load_frag_b(Bb, (grp * 4 + j) * 16, lane);
#pragma unroll
      for (int j = 0; j < 4; ++j)
        acc[grp * 4 + j] = __builtin_amdgcn_wmma_f32_16x16x32_f16(false, a, false, bf[j], (short)0,
                                                                  acc[grp * 4 + j], false, false);
    }
    sts(buf ^ 1);
    SGB(SG_VMREAD, 4);
    SGB(SG_DSREAD, 18);
    SGB(SG_MFMA, 4);
    SGB(SG_DSWRITE, 4);
    SGB(SG_MFMA, 4);
    __syncthreads();
    buf ^= 1;
  }

  const int g = lane >> 4, nr = lane & 15;
#pragma unroll
  for (int nt = 0; nt < 4; ++nt) {
    const int h = nt * 16 + nr;
#pragma unroll
    for (int r = 0; r < 8; ++r) {
      int trow = M0 + wave * 16 + g * 8 + r;
      size_t grow = (size_t)b * TSZ + trow;
      float num = acc[nt][r], den = acc[nt + 4][r];
      float y = sigQ[grow * HSZ + h] * (num / den);
      Yth[grow * HSZ + h] = (_Float16)y;
    }
  }
}

// ---------------- GEMM 3: output projection ----------------
// out[32768 x 512] = Yth[32768 x 64] @ wp^T (B(k=h,n=d)=wp[d][h]) + bp
__global__ __launch_bounds__(256) void k_out_gemm(
    const _Float16* __restrict__ Yth, const _Float16* __restrict__ wph,
    const float* __restrict__ bp, float* __restrict__ out) {
  __shared__ _Float16 As[128 * APAD];
  __shared__ _Float16 Bs[128 * APAD];
  const int tid = threadIdx.x, wave = tid >> 5, lane = tid & 31;
  const int M0 = blockIdx.x * 128;
  const int N0 = blockIdx.y * 128;
  const int arow = tid >> 1, ap16 = (tid & 1) << 4;
  v8f acc[8] = {};
  for (int K0 = 0; K0 < HSZ; K0 += 32) {
    __syncthreads();
    {
      const uint4* pa = reinterpret_cast<const uint4*>(Yth + (size_t)(M0 + arow) * HSZ + K0 + ap16);
      uint4 a0 = pa[0], a1 = pa[1];
      const uint4* pb = reinterpret_cast<const uint4*>(wph + (size_t)(N0 + arow) * HSZ + K0 + ap16);
      uint4 b0 = pb[0], b1 = pb[1];
      uint4* da = reinterpret_cast<uint4*>(&As[arow * APAD + ap16]);
      da[0] = a0; da[1] = a1;
      uint4* db = reinterpret_cast<uint4*>(&Bs[arow * APAD + ap16]);
      db[0] = b0; db[1] = b1;
    }
    __syncthreads();
    v16h a = load_frag_a(As, wave * 16, lane);
#pragma unroll
    for (int grp = 0; grp < 2; ++grp) {
      v16h bf[4];
#pragma unroll
      for (int j = 0; j < 4; ++j) bf[j] = load_frag_b(Bs, (grp * 4 + j) * 16, lane);
#pragma unroll
      for (int j = 0; j < 4; ++j)
        acc[grp * 4 + j] = __builtin_amdgcn_wmma_f32_16x16x32_f16(false, a, false, bf[j], (short)0,
                                                                  acc[grp * 4 + j], false, false);
    }
    SGB(SG_DSREAD, 18);
    SGB(SG_MFMA, 8);
  }
  const int g = lane >> 4, nr = lane & 15;
#pragma unroll
  for (int nt = 0; nt < 8; ++nt) {
    const int col = N0 + nt * 16 + nr;
    const float bpv = bp[col];
#pragma unroll
    for (int r = 0; r < 8; ++r) {
      int row = M0 + wave * 16 + g * 8 + r;
      out[(size_t)row * DSZ + col] = acc[nt][r] + bpv;
    }
  }
}

extern "C" void kernel_launch(void* const* d_in, const int* in_sizes, int n_in,
                              void* d_out, int out_size, void* d_ws, size_t ws_size,
                              hipStream_t stream) {
  (void)in_sizes; (void)n_in; (void)out_size; (void)ws_size;
  const float* x     = (const float*)d_in[0];
  const float* wq    = (const float*)d_in[1];
  const float* bq    = (const float*)d_in[2];
  const float* wk    = (const float*)d_in[3];
  const float* bk    = (const float*)d_in[4];
  const float* wv    = (const float*)d_in[5];
  const float* bv    = (const float*)d_in[6];
  const float* wp    = (const float*)d_in[7];
  const float* bp    = (const float*)d_in[8];
  const float* wbias = (const float*)d_in[9];
  float* out = (float*)d_out;

  char* ws = (char*)d_ws;
  size_t off = 0;
  auto alloc = [&](size_t bytes) -> char* {
    char* p = ws + off;
    off += (bytes + 255) & ~(size_t)255;
    return p;
  };
  _Float16* xh    = (_Float16*)alloc((size_t)MSZ * DSZ * 2);       // 32 MB
  _Float16* wqkvT = (_Float16*)alloc((size_t)192 * DSZ * 2);       // 192 KB, [192][512] row-major
  _Float16* ewh   = (_Float16*)alloc((size_t)TSZ * TSZ * 2);       // 32 MB
  _Float16* wph   = (_Float16*)alloc((size_t)DSZ * HSZ * 2);       // 64 KB
  float*    sigQ  = (float*)   alloc((size_t)MSZ * HSZ * 4);       // 8 MB
  _Float16* Zt    = (_Float16*)alloc((size_t)BSZ * 128 * TSZ * 2); // 8 MB, [b*128+h'][T]
  _Float16* Yth   = (_Float16*)alloc((size_t)MSZ * HSZ * 2);       // 4 MB

  k_cast_f32_f16<<<2048, 256, 0, stream>>>(x, xh, (MSZ * DSZ) / 4);
  // wqkvT = concat(wq, wk, wv) rows, cast to f16 (already [192][512] row-major)
  k_cast_f32_f16<<<32, 256, 0, stream>>>(wq, wqkvT,                 (HSZ * DSZ) / 4);
  k_cast_f32_f16<<<32, 256, 0, stream>>>(wk, wqkvT + HSZ * DSZ,     (HSZ * DSZ) / 4);
  k_cast_f32_f16<<<32, 256, 0, stream>>>(wv, wqkvT + 2 * HSZ * DSZ, (HSZ * DSZ) / 4);
  k_exp_f32_f16<<<4096, 256, 0, stream>>>(wbias, ewh, (TSZ * TSZ) / 4);
  k_cast_f32_f16<<<32, 256, 0, stream>>>(wp, wph, (DSZ * HSZ) / 4);

  k_qkv_gemm<<<dim3(MSZ / 128), 256, 0, stream>>>(xh, wqkvT, bq, bk, bv, sigQ, Zt);
  k_aft_core<<<dim3(TSZ / 128, BSZ), 256, 0, stream>>>(ewh, Zt, sigQ, Yth);
  k_out_gemm<<<dim3(MSZ / 128, DSZ / 128), 256, 0, stream>>>(Yth, wph, bp, out);
}
